// PrototypeLayer_13400297963666
// MI455X (gfx1250) — matrix operand order repaired and checked by reading, not compiled, not run on hardware
//
#include <hip/hip_runtime.h>
#include <hip/hip_bf16.h>

// ---------------- problem constants ----------------
#define NB    256   // batch
#define CIN   256   // input channels
#define HWSZ  784   // 28*28
#define DD    64    // prototype channel dim
#define PP    200   // num prototypes
#define PPAD  208   // padded to 13 WMMA col tiles
#define NCLS  10
#define MT    112   // spatial rows per block (784 = 7*112)
#define TPB   256   // 8 waves of 32

typedef __attribute__((ext_vector_type(16))) __bf16 v16bf;
typedef __attribute__((ext_vector_type(8)))  float  v8f;

union ABf { unsigned u[8]; v16bf v; };
union CDf { float f[8]; v8f v; };

__device__ __forceinline__ unsigned short f2bf(float f) {
  unsigned u = __float_as_uint(f);
  u += 0x7FFFu + ((u >> 16) & 1u);   // round-to-nearest-even (prep kernel)
  return (unsigned short)(u >> 16);
}
__device__ __forceinline__ float bf2f(unsigned short h) {
  return __uint_as_float(((unsigned)h) << 16);
}
// single-instruction packed f32->bf16 (low = a, high = b)
__device__ __forceinline__ unsigned cvt_pk_bf16(float a, float b) {
  unsigned r;
  asm("v_cvt_pk_bf16_f32 %0, %1, %2" : "=v"(r) : "v"(a), "v"(b));
  return r;
}
__device__ __forceinline__ unsigned short f2bf_hw(float a) {
  return (unsigned short)(cvt_pk_bf16(a, a) & 0xFFFFu);
}

// ---------------- prep: bf16 conversion, ||p||^2, min-buffer init ----------------
__global__ void prep_kernel(const float* __restrict__ W1, const float* __restrict__ W2,
                            const float* __restrict__ proto,
                            unsigned short* __restrict__ w1bf, unsigned short* __restrict__ w2bf,
                            unsigned short* __restrict__ pbf, float* __restrict__ p2,
                            unsigned* __restrict__ minbits) {
  int i = blockIdx.x * blockDim.x + threadIdx.x;
  const int s1 = DD * CIN;      // 16384
  const int s2 = DD * DD;       // 4096
  const int s3 = PPAD * DD;     // 13312
  const int s4 = PPAD;          // 208
  const int s5 = NB * PP;       // 51200
  if (i < s1) { w1bf[i] = f2bf(W1[i]); return; }
  i -= s1;
  if (i < s2) { w2bf[i] = f2bf(W2[i]); return; }
  i -= s2;
  if (i < s3) { int p = i / DD; pbf[i] = (p < PP) ? f2bf(proto[i]) : (unsigned short)0; return; }
  i -= s3;
  if (i < s4) {
    float s = 0.f;
    if (i < PP) {
      for (int d = 0; d < DD; ++d) { float v = proto[i * DD + d]; s += v * v; }
    }
    p2[i] = s; return;
  }
  i -= s4;
  if (i < s5) { minbits[i] = 0x7F7FFFFFu; }  // FLT_MAX bits
}

// ---------------- fused: conv1x1->relu->conv1x1->sigmoid->L2 dist->min ----------------
//
// LDS overlay plan (50,176 B total):
//   [0      , 28672) : xaf  float[2][3584]  x-chunk double buffer (stage 1, async copies)
//                      fa   bf16 [112*64]   f matrix (stages 2/3, after xaf dies)
//   [28672  , 35840) : xa   bf16 [112*32]   converted A-chunk (stage 1)
//                      f2s  f32  [112] + dmin u32[208] (stage 3, after xa dies)
//   [35840  , 50176) : ha   bf16 [112*64]   h matrix
__global__ __launch_bounds__(TPB)
void fused_kernel(const float* __restrict__ x,
                  const float* __restrict__ b1v,
                  const float* __restrict__ b2v,
                  const unsigned short* __restrict__ w1bf,
                  const unsigned short* __restrict__ w2bf,
                  const unsigned short* __restrict__ pbf,
                  const float* __restrict__ p2,
                  unsigned* __restrict__ minbits) {
  __shared__ __align__(16) char smem[50176];
  float*          xaf  = (float*)smem;                       // 2 x 3584 floats
  unsigned short* fa   = (unsigned short*)smem;              // 112*64 bf16 (overlay)
  unsigned short* xa   = (unsigned short*)(smem + 28672);    // 112*32 bf16
  float*          f2s  = (float*)(smem + 28672);             // 112 f32   (overlay)
  unsigned*       dmin = (unsigned*)(smem + 28672 + 448);    // 208 u32   (overlay)
  unsigned short* ha   = (unsigned short*)(smem + 35840);    // 112*64 bf16

  const int tid   = threadIdx.x;
  const int wave  = tid >> 5;          // wave32
  const int lane  = tid & 31;
  const int lhalf = lane >> 4;         // 0/1
  const int l16   = lane & 15;

  const int blk = blockIdx.x;
  const int n   = blk / 7;
  const int hw0 = (blk % 7) * MT;

  const float* xbase = x + (size_t)n * CIN * HWSZ + hw0;

  // Issue async global->LDS copy of one 112x32 fp32 chunk (channel-major in LDS).
  // Each lane copies an 8-byte pair of consecutive spatial elements -> contiguous
  // in both global memory and LDS. 7 issues x 256 threads x 8B = 14,336 B.
  auto issue_chunk = [&](int kc, int buf) {
    const float* gp = xbase + (size_t)(kc * 32) * HWSZ;
    #pragma unroll
    for (int it = 0; it < 7; ++it) {
      int i  = (it * TPB + tid) * 2;          // paired element index, 0..3582
      int cl = i / MT, m = i - cl * MT;       // channel-in-chunk, spatial row
      unsigned lds = (unsigned)(uintptr_t)(xaf + buf * 3584 + cl * MT + m);
      unsigned long long ga = (unsigned long long)(uintptr_t)(gp + (size_t)cl * HWSZ + m);
      asm volatile("global_load_async_to_lds_b64 %0, %1, off"
                   :: "v"(lds), "v"(ga) : "memory");
    }
  };

  // ======== Stage 1: h = relu(x @ W1^T + b1), M=112 K=256 Nout=64 ========
  CDf acc1[4];
  #pragma unroll
  for (int t = 0; t < 4; ++t) { v8f z = {}; acc1[t].v = z; }

  issue_chunk(0, 0);                           // prefetch chunk 0

  for (int kc = 0; kc < 8; ++kc) {
    const int buf = kc & 1;
    asm volatile("s_wait_asynccnt 0x0" ::: "memory");  // chunk kc landed in LDS
    __syncthreads();                                   // whole block sees it; xa free
    if (kc < 7) issue_chunk(kc + 1, buf ^ 1);          // overlap next copy with compute

    // repack fp32 chunk -> bf16 A-layout: xa[m][cl].
    // Two phases (all loads, then all cvt+stores) so the compiler can clause the
    // LDS loads instead of serializing on may-alias stores.
    {
      float ra[7], rb[7];
      #pragma unroll
      for (int it = 0; it < 7; ++it) {
        int i  = it * TPB + tid;               // 0..1791 : (channel pair, row)
        int cp = i / MT, m = i - cp * MT;
        ra[it] = xaf[buf * 3584 + (2 * cp) * MT + m];
        rb[it] = xaf[buf * 3584 + (2 * cp + 1) * MT + m];
      }
      #pragma unroll
      for (int it = 0; it < 7; ++it) {
        int i  = it * TPB + tid;
        int cp = i / MT, m = i - cp * MT;
        *(unsigned*)&xa[m * 32 + 2 * cp] = cvt_pk_bf16(ra[it], rb[it]);
      }
    }
    __syncthreads();

    #pragma unroll
    for (int ti = 0; ti < 4; ++ti) {
      int t = wave * 4 + ti; if (t > 27) t = 27;       // clamp: benign duplicate
      int rt = t >> 2, ct = t & 3;
      ABf A, B;
      #pragma unroll
      for (int j = 0; j < 8; ++j) {
        int ka = 2 * j + ((j >= 4) ? 8 : 0) + 8 * lhalf;   // A layout 16x32 bf16
        A.u[j] = *(const unsigned*)&xa[(rt * 16 + l16) * 32 + ka];
        int kb = 2 * j + ((j >= 4) ? 8 : 0) + 8 * lhalf;   // B layout 32x16 bf16
        B.u[j] = *(const unsigned*)&w1bf[(ct * 16 + l16) * CIN + kc * 32 + kb];
      }
      acc1[ti].v = __builtin_amdgcn_wmma_f32_16x16x32_bf16(
          false, A.v, false, B.v, (short)0, acc1[ti].v, false, false);
    }
  }
  __syncthreads();
  #pragma unroll
  for (int ti = 0; ti < 4; ++ti) {
    int t = wave * 4 + ti; if (t > 27) t = 27;
    int rt = t >> 2, ct = t & 3;
    int col = ct * 16 + l16;
    float bias = b1v[col];
    #pragma unroll
    for (int r = 0; r < 8; ++r) {
      int row = rt * 16 + r + 8 * lhalf;                  // C/D layout
      float hv = acc1[ti].f[r] + bias;
      hv = hv > 0.f ? hv : 0.f;
      ha[row * DD + col] = f2bf_hw(hv);
    }
  }
  __syncthreads();                            // ha ready; xaf region now dead -> fa

  // ======== Stage 2: f = sigmoid(h @ W2^T + b2), K=64 ========
  #pragma unroll
  for (int ti = 0; ti < 4; ++ti) {
    int t = wave * 4 + ti; if (t > 27) t = 27;
    int rt = t >> 2, ct = t & 3;
    CDf acc; { v8f z = {}; acc.v = z; }
    #pragma unroll
    for (int kc = 0; kc < 2; ++kc) {
      ABf A, B;
      #pragma unroll
      for (int j = 0; j < 8; ++j) {
        int ka = 2 * j + ((j >= 4) ? 8 : 0) + 8 * lhalf;
        A.u[j] = *(const unsigned*)&ha[(rt * 16 + l16) * DD + kc * 32 + ka];
        int kb = 2 * j + ((j >= 4) ? 8 : 0) + 8 * lhalf;
        B.u[j] = *(const unsigned*)&w2bf[(ct * 16 + l16) * DD + kc * 32 + kb];
      }
      acc.v = __builtin_amdgcn_wmma_f32_16x16x32_bf16(
          false, A.v, false, B.v, (short)0, acc.v, false, false);
    }
    int col = ct * 16 + l16;
    float bias = b2v[col];
    #pragma unroll
    for (int r = 0; r < 8; ++r) {
      int row = rt * 16 + r + 8 * lhalf;
      float fv = acc.f[r] + bias;
      fv = 1.0f / (1.0f + __expf(-fv));
      fa[row * DD + col] = f2bf_hw(fv);
    }
  }
  __syncthreads();                            // fa complete; xa region dead -> f2s/dmin
  if (tid < PPAD) dmin[tid] = 0x7F7FFFFFu;
  if (tid < MT) {
    float s = 0.f;
    #pragma unroll
    for (int c2 = 0; c2 < DD / 2; ++c2) {
      unsigned u = *(const unsigned*)&fa[tid * DD + 2 * c2];
      float a = bf2f((unsigned short)(u & 0xFFFFu));
      float b = bf2f((unsigned short)(u >> 16));
      s += a * a + b * b;
    }
    f2s[tid] = s;
  }
  __syncthreads();                            // f2s / dmin ready

  // ======== Stage 3: xp = f @ proto^T ; dist ; min ========
  #pragma unroll
  for (int ti = 0; ti < 12; ++ti) {
    int t = wave * 12 + ti; if (t > 90) t = 90;          // 91 real tiles (7x13)
    int rt = t / 13, ct = t % 13;
    CDf acc; { v8f z = {}; acc.v = z; }
    #pragma unroll
    for (int kc = 0; kc < 2; ++kc) {
      ABf A, B;
      #pragma unroll
      for (int j = 0; j < 8; ++j) {
        int ka = 2 * j + ((j >= 4) ? 8 : 0) + 8 * lhalf;
        A.u[j] = *(const unsigned*)&fa[(rt * 16 + l16) * DD + kc * 32 + ka];
        int kb = 2 * j + ((j >= 4) ? 8 : 0) + 8 * lhalf;
        B.u[j] = *(const unsigned*)&pbf[(ct * 16 + l16) * DD + kc * 32 + kb];
      }
      acc.v = __builtin_amdgcn_wmma_f32_16x16x32_bf16(
          false, A.v, false, B.v, (short)0, acc.v, false, false);
    }
    int col = ct * 16 + l16;
    float pc = p2[col];
    #pragma unroll
    for (int r = 0; r < 8; ++r) {
      int row = rt * 16 + r + 8 * lhalf;
      float d = f2s[row] - 2.0f * acc.f[r] + pc;
      d = d > 0.f ? d : 0.f;                             // relu => d >= 0
      if (col < PP) atomicMin(&dmin[col], __float_as_uint(d));
    }
  }
  __syncthreads();
  if (tid < PP) atomicMin(&minbits[n * PP + tid], dmin[tid]);
}

// ---------------- epilogue: logits = -min_d @ last_W^T + last_b ----------------
__global__ void logits_kernel(const float* __restrict__ mind,
                              const float* __restrict__ lastW,
                              const float* __restrict__ lastb,
                              float* __restrict__ logits) {
  int i = blockIdx.x * blockDim.x + threadIdx.x;
  if (i >= NB * NCLS) return;
  int n = i / NCLS, c = i % NCLS;
  float s = lastb[c];
  for (int p = 0; p < PP; ++p) s -= mind[n * PP + p] * lastW[c * PP + p];
  logits[i] = s;
}

extern "C" void kernel_launch(void* const* d_in, const int* in_sizes, int n_in,
                              void* d_out, int out_size, void* d_ws, size_t ws_size,
                              hipStream_t stream) {
  const float* x    = (const float*)d_in[0];
  const float* W1   = (const float*)d_in[1];
  const float* b1   = (const float*)d_in[2];
  const float* W2   = (const float*)d_in[3];
  const float* b2   = (const float*)d_in[4];
  const float* prot = (const float*)d_in[5];
  const float* lW   = (const float*)d_in[6];
  const float* lb   = (const float*)d_in[7];

  float* out    = (float*)d_out;
  float* logits = out;              // (N, NCLS) = 2560 floats
  float* mind   = out + NB * NCLS;  // (N, P)    = 51200 floats (uint bits during min)

  unsigned short* w1bf = (unsigned short*)d_ws;            // 16384 bf16
  unsigned short* w2bf = w1bf + DD * CIN;                  // 4096  bf16
  unsigned short* pbf  = w2bf + DD * DD;                   // 13312 bf16 (padded)
  float*          p2   = (float*)(pbf + PPAD * DD);        // 208 f32 (4B aligned)

  const int prepN = DD * CIN + DD * DD + PPAD * DD + PPAD + NB * PP;
  hipLaunchKernelGGL(prep_kernel, dim3((prepN + 255) / 256), dim3(256), 0, stream,
                     W1, W2, prot, w1bf, w2bf, pbf, p2, (unsigned*)mind);

  hipLaunchKernelGGL(fused_kernel, dim3(NB * 7), dim3(TPB), 0, stream,
                     x, b1, b2, w1bf, w2bf, pbf, p2, (unsigned*)mind);

  hipLaunchKernelGGL(logits_kernel, dim3((NB * NCLS + 255) / 256), dim3(256), 0, stream,
                     mind, lW, lb, logits);
}
